// FUNASR_NANO_DECODER_MAIN_8091718386205
// MI455X (gfx1250) — matrix operand order repaired and checked by reading, not compiled, not run on hardware
//
#include <hip/hip_runtime.h>
#include <hip/hip_bf16.h>
#include <stdint.h>

// ---------------------------------------------------------------------------
// Model constants (fixed problem from the reference)
// ---------------------------------------------------------------------------
#define S_LEN   1024
#define D_MODEL 1024
#define NHEADS  16
#define NKVH    8
#define HDIM    128
#define DFF     3072
#define NLAYERS 8
#define VOCAB   32000
#define O_QKV   4096        // (16 + 2*8) * 128
#define BK      64          // GEMM K-tile (2 WMMA K-steps per stage)

typedef __attribute__((ext_vector_type(16))) __bf16 v16bf;
typedef __attribute__((ext_vector_type(8)))  __bf16 v8bf;
typedef __attribute__((ext_vector_type(8)))  float  v8f;
typedef int v4i_t __attribute__((vector_size(16)));   // matches builtin param

#define DEV __device__ __forceinline__

// ---------------------------------------------------------------------------
// CDNA5 async global->LDS copy (GLOBAL_LOAD_ASYNC_TO_LDS_B128, ASYNCcnt),
// with a synchronous fallback if the builtin is unavailable.
// Signature (from compiler diagnostic): (v4i AS1*, v4i AS3*, imm off, imm cpol)
// ---------------------------------------------------------------------------
#if defined(__AMDGCN__) && __has_builtin(__builtin_amdgcn_global_load_async_to_lds_b128)
#define HAVE_ASYNC_LDS 1
#else
#define HAVE_ASYNC_LDS 0
#endif

DEV void async_cp16(const void* g, void* l) {
#if HAVE_ASYNC_LDS
    __builtin_amdgcn_global_load_async_to_lds_b128(
        (__attribute__((address_space(1))) v4i_t*)(uintptr_t)g,
        (__attribute__((address_space(3))) v4i_t*)(unsigned int)(uintptr_t)l,
        0, 0);
#else
    *(v8bf*)l = *(const v8bf*)g;
#endif
}

DEV void async_wait0() {
#if HAVE_ASYNC_LDS
#if __has_builtin(__builtin_amdgcn_s_wait_asynccnt)
    __builtin_amdgcn_s_wait_asynccnt(0);
#endif
#endif
}

// ---------------------------------------------------------------------------
// Generic bf16 WMMA GEMM:  C[M,N] = A[M,K] (bf16, row-major)  x  B
//   BMODE 0: B is fp32, row-major (N,K)   -> converted to bf16 per tile
//   BMODE 1: B is bf16, row-major (N,K)   -> async staged
//   BMODE 2: B is bf16, row-major (K,N)   -> transposed into LDS
// Batched along blockIdx.z; B batch index is (z / bBatchDiv) for GQA.
// Tile: 128x128x64, double-buffered LDS, 256 threads = 8 waves (2x4 grid),
// each wave: 4(M) x 2(N) fragments, 16 v_wmma_f32_16x16x32_bf16 per stage.
// ---------------------------------------------------------------------------
template <int BMODE>
__global__ __launch_bounds__(256) void gemm_bf16_wmma(
    const __bf16* __restrict__ A, const void* __restrict__ Bv,
    float* __restrict__ C, int M, int N, int K, int lda, int ldb, int ldc,
    long aBatchStride, long bBatchStride, long cBatchStride, int bBatchDiv)
{
    __shared__ __bf16 As[2][128][BK];   // [buf][m][k]   16KB each
    __shared__ __bf16 Bs[2][128][BK];   // [buf][n][k]   16KB each

    const int tid  = threadIdx.x;
    const int lane = tid & 31;
    const int wave = tid >> 5;
    const int wm   = wave >> 2;      // 0..1  -> 64-row slab
    const int wn   = wave & 3;       // 0..3  -> 32-col slab
    const int hp   = lane >> 4;      // lane half (0/1)
    const int l16  = lane & 15;

    const int m0 = blockIdx.y * 128;
    const int n0 = blockIdx.x * 128;
    const int b  = blockIdx.z;

    A += (long)b * aBatchStride;
    C += (long)b * cBatchStride;
    const float*  Bf = (const float*) Bv + (long)(b / bBatchDiv) * bBatchStride;
    const __bf16* Bh = (const __bf16*)Bv + (long)(b / bBatchDiv) * bBatchStride;
    (void)M;

    v8f acc[4][2];
#pragma unroll
    for (int i = 0; i < 4; ++i)
#pragma unroll
        for (int j = 0; j < 2; ++j) acc[i][j] = {};

    // ---- tile staging -----------------------------------------------------
    auto stageA = [&](int k0, int buf) {
        const int r = tid >> 1;
        const int c = (tid & 1) * 32;               // 32 bf16 = 64B per thread
        const __bf16* src = A + (long)(m0 + r) * lda + (k0 + c);
        __bf16* dst = &As[buf][r][c];
#pragma unroll
        for (int i = 0; i < 32; i += 8) async_cp16(src + i, dst + i);
    };

    auto stageB = [&](int k0, int buf) {
        if (BMODE == 0) {                           // fp32 (N,K) -> bf16
            const int n = tid >> 1;
            const int c = (tid & 1) * 32;
            const float* src = Bf + (long)(n0 + n) * ldb + (k0 + c);
            __bf16 tmp[32];
#pragma unroll
            for (int i = 0; i < 32; i += 4) {
                float4 f = *(const float4*)(src + i);
                tmp[i + 0] = (__bf16)f.x;
                tmp[i + 1] = (__bf16)f.y;
                tmp[i + 2] = (__bf16)f.z;
                tmp[i + 3] = (__bf16)f.w;
            }
#pragma unroll
            for (int i = 0; i < 32; i += 8)
                *(v8bf*)(&Bs[buf][n][c + i]) = *(const v8bf*)(&tmp[i]);
            __builtin_prefetch((const void*)(src + BK), 0, 1); // global_prefetch_b8
        } else if (BMODE == 1) {                    // bf16 (N,K) async
            const int n = tid >> 1;
            const int c = (tid & 1) * 32;
            const __bf16* src = Bh + (long)(n0 + n) * ldb + (k0 + c);
            __bf16* dst = &Bs[buf][n][c];
#pragma unroll
            for (int i = 0; i < 32; i += 8) async_cp16(src + i, dst + i);
        } else {                                    // bf16 (K,N): transpose
            const int kk = tid >> 2;                // 0..63
            const int n  = (tid & 3) * 32;          // 0,32,64,96
            const __bf16* src = Bh + (long)(k0 + kk) * ldb + (n0 + n);
#pragma unroll
            for (int i = 0; i < 32; ++i) Bs[buf][n + i][kk] = src[i];
        }
    };

    // ---- double-buffered main loop ----------------------------------------
    const int nk = K / BK;
    stageA(0, 0);
    stageB(0, 0);

    for (int kt = 0; kt < nk; ++kt) {
        const int cur = kt & 1;
        async_wait0();                 // this wave's async tile is in LDS
        __syncthreads();               // all waves' tiles visible
        if (kt + 1 < nk) {             // overlap next-tile staging with WMMA
            stageA((kt + 1) * BK, cur ^ 1);
            stageB((kt + 1) * BK, cur ^ 1);
        }

#pragma unroll
        for (int ks = 0; ks < BK; ks += 32) {
            // B fragments (32x16): lane N=l16, K chunks [hp*8..+7],[16+hp*8..+7]
            v16bf bfrag[2];
#pragma unroll
            for (int nf = 0; nf < 2; ++nf) {
                const int n = wn * 32 + nf * 16 + l16;
                v8bf lo = *(const v8bf*)(&Bs[cur][n][ks + hp * 8]);
                v8bf hi = *(const v8bf*)(&Bs[cur][n][ks + 16 + hp * 8]);
#pragma unroll
                for (int i = 0; i < 8; ++i) { bfrag[nf][i] = lo[i]; bfrag[nf][i + 8] = hi[i]; }
            }
            // A fragments (16x32): lane row=l16, K chunks [hp*8..+7],[16+hp*8..+7]
#pragma unroll
            for (int mf = 0; mf < 4; ++mf) {
                const int r = wm * 64 + mf * 16 + l16;
                v8bf lo = *(const v8bf*)(&As[cur][r][ks + hp * 8]);
                v8bf hi = *(const v8bf*)(&As[cur][r][ks + 16 + hp * 8]);
                v16bf afrag;
#pragma unroll
                for (int i = 0; i < 8; ++i) { afrag[i] = lo[i]; afrag[i + 8] = hi[i]; }
#pragma unroll
                for (int nf = 0; nf < 2; ++nf) {
                    acc[mf][nf] = __builtin_amdgcn_wmma_f32_16x16x32_bf16(
                        false, afrag, false, bfrag[nf], (short)0, acc[mf][nf],
                        false, false);
                }
            }
        }
        __syncthreads();               // compute done before buffer reuse
    }

    // ---- epilogue: C frag VGPR i -> row = base + i + hp*8, col = l16 ------
#pragma unroll
    for (int mf = 0; mf < 4; ++mf) {
#pragma unroll
        for (int nf = 0; nf < 2; ++nf) {
            const int rbase = m0 + wm * 64 + mf * 16 + hp * 8;
            const int col   = n0 + wn * 32 + nf * 16 + l16;
#pragma unroll
            for (int i = 0; i < 8; ++i)
                C[(long)(rbase + i) * ldc + col] = acc[mf][nf][i];
        }
    }
}

// ---------------------------------------------------------------------------
// Reductions
// ---------------------------------------------------------------------------
DEV float waveSum(float v) {
#pragma unroll
    for (int off = 16; off; off >>= 1) v += __shfl_down(v, off, 32);
    return v;
}
DEV float waveMax(float v) {
#pragma unroll
    for (int off = 16; off; off >>= 1) v = fmaxf(v, __shfl_down(v, off, 32));
    return v;
}
DEV float blockSum128(float v, float* red) {
    v = waveSum(v);
    if ((threadIdx.x & 31) == 0) red[threadIdx.x >> 5] = v;
    __syncthreads();
    float t = red[0] + red[1] + red[2] + red[3];
    __syncthreads();
    return t;
}
DEV float blockSum256(float v, float* red) {
    v = waveSum(v);
    if ((threadIdx.x & 31) == 0) red[threadIdx.x >> 5] = v;
    __syncthreads();
    float t = 0.f;
#pragma unroll
    for (int i = 0; i < 8; ++i) t += red[i];
    __syncthreads();
    return t;
}
DEV float blockMax256(float v, float* red) {
    v = waveMax(v);
    if ((threadIdx.x & 31) == 0) red[threadIdx.x >> 5] = v;
    __syncthreads();
    float t = red[0];
#pragma unroll
    for (int i = 1; i < 8; ++i) t = fmaxf(t, red[i]);
    __syncthreads();
    return t;
}

// ---------------------------------------------------------------------------
// hn = x * rsqrt(sum(x^2)) per row, output bf16 (for GEMM input)
// ---------------------------------------------------------------------------
__global__ __launch_bounds__(256) void l2norm_bf16_kernel(
    const float* __restrict__ h, __bf16* __restrict__ out)
{
    __shared__ float red[8];
    const int s = blockIdx.x;
    const float* row = h + (long)s * D_MODEL;
    float ss = 0.f;
    for (int d = threadIdx.x; d < D_MODEL; d += 256) { float v = row[d]; ss += v * v; }
    float tot = blockSum256(ss, red);
    float sc = rsqrtf(tot);
    for (int d = threadIdx.x; d < D_MODEL; d += 256)
        out[(long)s * D_MODEL + d] = (__bf16)(row[d] * sc);
}

// last-row l2 scale, fp32 output (LM head input)
__global__ __launch_bounds__(256) void lastrow_l2norm_kernel(
    const float* __restrict__ h, float* __restrict__ hl)
{
    __shared__ float red[8];
    const float* row = h + (long)(S_LEN - 1) * D_MODEL;
    float ss = 0.f;
    for (int d = threadIdx.x; d < D_MODEL; d += 256) { float v = row[d]; ss += v * v; }
    float tot = blockSum256(ss, red);
    float sc = rsqrtf(tot);
    for (int d = threadIdx.x; d < D_MODEL; d += 256) hl[d] = row[d] * sc;
}

// ---------------------------------------------------------------------------
// QKV post: RMSNorm(q,k) (SCALE folded into norm weights) + RoPE,
// V pass-through; all converted to bf16 in [head][s][hd] layouts.
// grid = (S, 32): jobs 0..15 Q heads, 16..23 K heads, 24..31 V heads.
// ---------------------------------------------------------------------------
__global__ __launch_bounds__(128) void qkvpost_kernel(
    const float* __restrict__ qkv,
    const float* __restrict__ qw, const float* __restrict__ kw,
    const float* __restrict__ cosb, const float* __restrict__ sinb,
    __bf16* __restrict__ qb, __bf16* __restrict__ kb, __bf16* __restrict__ vb)
{
    __shared__ float red[4];
    __shared__ float vec[HDIM];
    const int s   = blockIdx.x;
    const int job = blockIdx.y;
    const int d   = threadIdx.x;   // 0..127

    if (job < 16) {                          // Q head
        const int hh = job;
        float x  = qkv[(long)s * O_QKV + hh * HDIM + d];
        float ms = blockSum128(x * x, red) * (1.0f / HDIM);
        float xn = x * rsqrtf(ms + 1e-6f) * qw[d];
        vec[d] = xn;
        __syncthreads();
        float rot = vec[(d + 64) & 127];     // rot_half (sin table pre-negated)
        float o   = xn * cosb[s * HDIM + d] + rot * sinb[s * HDIM + d];
        qb[((long)hh * S_LEN + s) * HDIM + d] = (__bf16)o;
    } else if (job < 24) {                   // K head
        const int kv = job - 16;
        float x  = qkv[(long)s * O_QKV + 2048 + kv * HDIM + d];
        float ms = blockSum128(x * x, red) * (1.0f / HDIM);
        float xn = x * rsqrtf(ms + 1e-6f) * kw[d];
        vec[d] = xn;
        __syncthreads();
        float rot = vec[(d + 64) & 127];
        float o   = xn * cosb[s * HDIM + d] + rot * sinb[s * HDIM + d];
        kb[((long)kv * S_LEN + s) * HDIM + d] = (__bf16)o;
    } else {                                 // V head
        const int kv = job - 24;
        float x = qkv[(long)s * O_QKV + 3072 + kv * HDIM + d];
        vb[((long)kv * S_LEN + s) * HDIM + d] = (__bf16)x;
    }
}

// ---------------------------------------------------------------------------
// Causal masked softmax over T=1024, fp32 scores -> bf16 probs.
// grid = (S, NHEADS), 256 threads, 4 elements/thread.
// ---------------------------------------------------------------------------
__global__ __launch_bounds__(256) void softmax_kernel(
    const float* __restrict__ scores, __bf16* __restrict__ probs,
    const float* __restrict__ mask)
{
    __shared__ float red[8];
    const int s = blockIdx.x;
    const int h = blockIdx.y;
    const float*  row  = scores + ((long)h * S_LEN + s) * S_LEN;
    __bf16*       prow = probs  + ((long)h * S_LEN + s) * S_LEN;
    const float mval = mask[0];

    float loc[4];
    float mx = -3.4e38f;
#pragma unroll
    for (int i = 0; i < 4; ++i) {
        int j = threadIdx.x + i * 256;
        float v = row[j] + ((j <= s) ? 0.f : -128.f * mval);
        loc[i] = v;
        mx = fmaxf(mx, v);
    }
    float gmx = blockMax256(mx, red);
    float sum = 0.f;
#pragma unroll
    for (int i = 0; i < 4; ++i) { loc[i] = __expf(loc[i] - gmx); sum += loc[i]; }
    float gsum = blockSum256(sum, red);
    float inv = 1.0f / gsum;
#pragma unroll
    for (int i = 0; i < 4; ++i)
        prow[threadIdx.x + i * 256] = (__bf16)(loc[i] * inv);
}

// ---------------------------------------------------------------------------
// Elementwise helpers
// ---------------------------------------------------------------------------
__global__ __launch_bounds__(256) void silu_mul_kernel(
    const float* __restrict__ gu, __bf16* __restrict__ act)
{
    long idx = (long)blockIdx.x * 256 + threadIdx.x;   // S*DFF total
    int sI = (int)(idx / DFF);
    int f  = (int)(idx % DFF);
    float g = gu[(long)sI * (2 * DFF) + f];
    float u = gu[(long)sI * (2 * DFF) + DFF + f];
    float si = g / (1.0f + __expf(-g));
    act[idx] = (__bf16)(si * u);
}

__global__ __launch_bounds__(256) void add_kernel(
    float* __restrict__ h, const float* __restrict__ d)
{
    long idx = (long)blockIdx.x * 256 + threadIdx.x;
    h[idx] += d[idx];
}

__global__ __launch_bounds__(256) void f32_to_bf16_kernel(
    const float* __restrict__ in, __bf16* __restrict__ out)
{
    long idx = (long)blockIdx.x * 256 + threadIdx.x;
    out[idx] = (__bf16)in[idx];
}

// ---------------------------------------------------------------------------
// LM head GEMV: logits[v] = dot(hl, W[v,:]); one wave per row.
// ---------------------------------------------------------------------------
__global__ __launch_bounds__(256) void lmhead_kernel(
    const float* __restrict__ hl, const float* __restrict__ W,
    float* __restrict__ out)
{
    const int wave = threadIdx.x >> 5;
    const int lane = threadIdx.x & 31;
    const int row  = blockIdx.x * 8 + wave;
    if (row >= VOCAB) return;
    const float* w = W + (long)row * D_MODEL;
    float sum = 0.f;
    for (int d = lane; d < D_MODEL; d += 32) sum += hl[d] * w[d];
    sum = waveSum(sum);
    if (lane == 0) out[row] = sum;
}

// ---------------------------------------------------------------------------
// Host-side orchestration
// ---------------------------------------------------------------------------
static inline size_t alignUp(size_t x) { return (x + 255) & ~(size_t)255; }

extern "C" void kernel_launch(void* const* d_in, const int* in_sizes, int n_in,
                              void* d_out, int out_size, void* d_ws, size_t ws_size,
                              hipStream_t stream)
{
    (void)in_sizes; (void)n_in; (void)out_size; (void)ws_size;

    const float* hs       = (const float*)d_in[0];
    const float* qkv_w    = (const float*)d_in[1];   // (8, 4096, 1024)
    const float* q_norm_w = (const float*)d_in[2];   // (8, 128)
    const float* k_norm_w = (const float*)d_in[3];   // (8, 128)
    const float* o_w      = (const float*)d_in[4];   // (8, 1024, 2048)
    const float* gu_w     = (const float*)d_in[5];   // (8, 6144, 1024)
    const float* dn_w     = (const float*)d_in[6];   // (8, 1024, 3072)
    const float* lm_w     = (const float*)d_in[7];   // (32000, 1024)
    const float* cos_tab  = (const float*)d_in[8];   // (2048, 128)
    const float* sin_tab  = (const float*)d_in[9];   // (2048, 128)
    const float* mask     = (const float*)d_in[12];  // (1,)
    float* logits = (float*)d_out;

    // ---- workspace layout
    char* base = (char*)d_ws;
    size_t off = 0;
    auto take = [&](size_t bytes) { void* p = base + off; off += alignUp(bytes); return p; };

    float*  hbuf  = (float*) take((size_t)S_LEN * D_MODEL * 4);          // running hidden
    __bf16* hn    = (__bf16*)take((size_t)S_LEN * D_MODEL * 2);
    float*  qkv   = (float*) take((size_t)S_LEN * O_QKV * 4);
    __bf16* qb    = (__bf16*)take((size_t)NHEADS * S_LEN * HDIM * 2);
    __bf16* kb    = (__bf16*)take((size_t)NKVH * S_LEN * HDIM * 2);
    __bf16* vb    = (__bf16*)take((size_t)NKVH * S_LEN * HDIM * 2);
    float*  scor  = (float*) take((size_t)NHEADS * S_LEN * S_LEN * 4);
    __bf16* probs = (__bf16*)take((size_t)NHEADS * S_LEN * S_LEN * 2);
    float*  attnf = (float*) take((size_t)S_LEN * (NHEADS * HDIM) * 4);
    __bf16* attnb = (__bf16*)take((size_t)S_LEN * (NHEADS * HDIM) * 2);
    float*  gu    = (float*) take((size_t)S_LEN * 2 * DFF * 4);
    __bf16* actb  = (__bf16*)take((size_t)S_LEN * DFF * 2);
    float*  delta = (float*) take((size_t)S_LEN * D_MODEL * 4);
    float*  hl    = (float*) take((size_t)D_MODEL * 4);

    (void)hipMemcpyAsync(hbuf, hs, (size_t)S_LEN * D_MODEL * 4,
                         hipMemcpyDeviceToDevice, stream);

    const dim3 blk256(256), blk128(128);

    for (int l = 0; l < NLAYERS; ++l) {
        // ---- attention block ------------------------------------------------
        l2norm_bf16_kernel<<<S_LEN, blk256, 0, stream>>>(hbuf, hn);

        // qkv = hn @ qkv_w[l]^T : (1024 x 4096), K=1024, B fp32 (N,K)
        gemm_bf16_wmma<0><<<dim3(O_QKV / 128, S_LEN / 128, 1), blk256, 0, stream>>>(
            hn, qkv_w + (size_t)l * O_QKV * D_MODEL, qkv,
            S_LEN, O_QKV, D_MODEL, D_MODEL, D_MODEL, O_QKV, 0, 0, 0, 1);

        qkvpost_kernel<<<dim3(S_LEN, 32), blk128, 0, stream>>>(
            qkv, q_norm_w + (size_t)l * HDIM, k_norm_w + (size_t)l * HDIM,
            cos_tab, sin_tab, qb, kb, vb);

        // scores[h] = q[h] @ k[h/2]^T : per head (1024 x 1024), K=128
        gemm_bf16_wmma<1><<<dim3(S_LEN / 128, S_LEN / 128, NHEADS), blk256, 0, stream>>>(
            qb, kb, scor,
            S_LEN, S_LEN, HDIM, HDIM, HDIM, S_LEN,
            (long)S_LEN * HDIM, (long)S_LEN * HDIM, (long)S_LEN * S_LEN, 2);

        softmax_kernel<<<dim3(S_LEN, NHEADS), blk256, 0, stream>>>(scor, probs, mask);

        // attn[h] = probs[h] @ v[h/2] : (1024 x 128), K=1024; B bf16 (K,N)
        // written into (S, 2048) at column h*128
        gemm_bf16_wmma<2><<<dim3(1, S_LEN / 128, NHEADS), blk256, 0, stream>>>(
            probs, vb, attnf,
            S_LEN, HDIM, S_LEN, S_LEN, HDIM, NHEADS * HDIM,
            (long)S_LEN * S_LEN, (long)S_LEN * HDIM, (long)HDIM, 2);

        f32_to_bf16_kernel<<<(S_LEN * NHEADS * HDIM) / 256, blk256, 0, stream>>>(attnf, attnb);

        // delta = attn @ o_w[l]^T : (1024 x 1024), K=2048
        gemm_bf16_wmma<0><<<dim3(D_MODEL / 128, S_LEN / 128, 1), blk256, 0, stream>>>(
            attnb, o_w + (size_t)l * D_MODEL * (NHEADS * HDIM), delta,
            S_LEN, D_MODEL, NHEADS * HDIM, NHEADS * HDIM, NHEADS * HDIM, D_MODEL,
            0, 0, 0, 1);

        add_kernel<<<(S_LEN * D_MODEL) / 256, blk256, 0, stream>>>(hbuf, delta);

        // ---- MLP block ------------------------------------------------------
        l2norm_bf16_kernel<<<S_LEN, blk256, 0, stream>>>(hbuf, hn);

        // gu = hn @ gate_up_w[l]^T : (1024 x 6144), K=1024
        gemm_bf16_wmma<0><<<dim3((2 * DFF) / 128, S_LEN / 128, 1), blk256, 0, stream>>>(
            hn, gu_w + (size_t)l * 2 * DFF * D_MODEL, gu,
            S_LEN, 2 * DFF, D_MODEL, D_MODEL, D_MODEL, 2 * DFF, 0, 0, 0, 1);

        silu_mul_kernel<<<(S_LEN * DFF) / 256, blk256, 0, stream>>>(gu, actb);

        // delta = act @ down_w[l]^T : (1024 x 1024), K=3072
        gemm_bf16_wmma<0><<<dim3(D_MODEL / 128, S_LEN / 128, 1), blk256, 0, stream>>>(
            actb, dn_w + (size_t)l * D_MODEL * DFF, delta,
            S_LEN, D_MODEL, DFF, DFF, DFF, D_MODEL, 0, 0, 0, 1);

        add_kernel<<<(S_LEN * D_MODEL) / 256, blk256, 0, stream>>>(hbuf, delta);
    }

    // ---- LM head ------------------------------------------------------------
    lastrow_l2norm_kernel<<<1, blk256, 0, stream>>>(hbuf, hl);
    lmhead_kernel<<<VOCAB / 8, blk256, 0, stream>>>(hl, lm_w, logits);
}
